// FullDimHybridBlock_76579266888380
// MI455X (gfx1250) — compile-verified, matrix operand order checked by reading, not run on hardware
//
#include <hip/hip_runtime.h>

typedef unsigned short u16;
typedef unsigned int   u32;
typedef unsigned long long u64;
typedef __attribute__((ext_vector_type(16))) __bf16 bf16x16;
typedef __attribute__((ext_vector_type(8)))  __bf16 bf16x8;
typedef __attribute__((ext_vector_type(8)))  float  f32x8;
typedef __attribute__((ext_vector_type(4)))  unsigned int v4u;
typedef __attribute__((ext_vector_type(8)))  int v8i;
typedef __attribute__((ext_vector_type(4)))  int v4i;

#define BQ   4
#define LQ   4096
#define DQ   1024
#define MQ   (BQ*LQ)
#define NSEG 16
#define SEGL (LQ/NSEG)
#define LDS_STRIDE 72   // 64 + 8 pad elems; rows stay 16B aligned (144B)

// ---------- helpers ----------
__device__ __forceinline__ u16 f2bf(float f) {
  u32 u = __float_as_uint(f);
  u += 0x7FFFu + ((u >> 16) & 1u);        // round-to-nearest-even
  return (u16)(u >> 16);
}

__device__ __forceinline__ float wave_sum(float v) {
#pragma unroll
  for (int off = 16; off; off >>= 1) v += __shfl_xor(v, off, 32);
  return v;
}

// ---------- f32 -> bf16 conversion (vectorized, 4/thread) ----------
__global__ void cvt_f32_bf16(const float* __restrict__ s, u16* __restrict__ d, int n4) {
  int i = blockIdx.x * 256 + threadIdx.x;
  if (i < n4) {
    float4 v = ((const float4*)s)[i];
    uint2 p;
    p.x = (u32)f2bf(v.x) | ((u32)f2bf(v.y) << 16);
    p.y = (u32)f2bf(v.z) | ((u32)f2bf(v.w) << 16);
    ((uint2*)d)[i] = p;
  }
}

// ---------- gate_in = concat(x, shift1(x)) in bf16 (8 elems/thread) ----------
__global__ void build_gate_in(const u16* __restrict__ xb, u16* __restrict__ gi) {
  size_t c = (size_t)blockIdx.x * 256 + threadIdx.x;   // 16B chunk id
  size_t i = c * 8;
  int m = (int)(i >> 11);
  int k = (int)(i & 2047);
  uint4 v;
  if (k < DQ) {
    v = *(const uint4*)(xb + (size_t)m * DQ + k);
  } else {
    int l = m & (LQ - 1);
    if (l == 0) { v.x = v.y = v.z = v.w = 0u; }
    else v = *(const uint4*)(xb + (size_t)(m - 1) * DQ + (k - DQ));
  }
  *(uint4*)(gi + i) = v;
}

// ---------- WMMA bf16 GEMM, 128x128 block tile ----------
// C[M,N] = act(A[M,K] @ W[K,N] + bias) (+res). A,W bf16 bits; C f32.
// M,N multiples of 128; K multiple of 64. mode: 0 none, 1 exact GELU.
__device__ __forceinline__ bf16x16 frag_from_lds(const u16* base, int rowbase, int ks) {
  const int lane = threadIdx.x & 31;
  const int half = lane >> 4;                    // lanes 0-15 vs 16-31
  const int r    = rowbase + (lane & 15);
  bf16x8 lo = *(const bf16x8*)(base + r * LDS_STRIDE + ks + half * 8);
  bf16x8 hi = *(const bf16x8*)(base + r * LDS_STRIDE + ks + 16 + half * 8);
  bf16x16 f;
#pragma unroll
  for (int i = 0; i < 8; i++) { f[i] = lo[i]; f[i + 8] = hi[i]; }
  return f;
}

#if __has_builtin(__builtin_amdgcn_tensor_load_to_lds) && __has_builtin(__builtin_amdgcn_s_wait_tensorcnt)
#define USE_TDM 1
#if defined(__clang_major__) && (__clang_major__ >= 23)
#define TDM_LOAD(g0, g1) __builtin_amdgcn_tensor_load_to_lds((g0), (g1), (v4i)0, (v4i)0, (v8i)0, 0)
#else
#define TDM_LOAD(g0, g1) __builtin_amdgcn_tensor_load_to_lds((g0), (g1), (v4i)0, (v4i)0, 0)
#endif
#else
#define USE_TDM 0
#endif

__global__ __launch_bounds__(256)
void gemm_bf16_wmma(const u16* __restrict__ A, const u16* __restrict__ Bw,
                    const float* __restrict__ bias, const float* __restrict__ res,
                    float* __restrict__ C, int M, int N, int K, int mode) {
  __shared__ __align__(16) u16 As[2][128 * LDS_STRIDE];   // A tile [m][k], stride 72
  __shared__ __align__(16) u16 Bs[2][128 * LDS_STRIDE];   // B tile TRANSPOSED [n][k]
  const int tid  = threadIdx.x;
  const int wave = tid >> 5;
  const int wr   = wave >> 1, wc = wave & 1;   // 4x2 wave grid; wave = 32(m) x 64(n)
  const int tileM = blockIdx.y * 128;
  const int tileN = blockIdx.x * 128;
  const u16* Aptr = A + (size_t)tileM * K;
  const u16* Bptr = Bw + tileN;

  f32x8 acc[2][4] = {};
  uint4 br[4];
#if !USE_TDM
  uint4 ar[4];
#endif

  // ---- staging helpers ----
  auto loadB = [&](int k0) {                       // B: 64(k) x 128(n)
#pragma unroll
    for (int i = 0; i < 4; i++) {
      int slot = tid + i * 256;                    // 0..1023, 8 n's per slot
      int k = slot >> 4, nc = slot & 15;
      br[i] = *(const uint4*)(Bptr + (size_t)(k0 + k) * N + nc * 8);
    }
  };
  auto storeB = [&](int buf) {
#pragma unroll
    for (int i = 0; i < 4; i++) {
      int slot = tid + i * 256;
      int k = slot >> 4, n0 = (slot & 15) * 8;
      u16 tmp[8]; *(uint4*)tmp = br[i];
#pragma unroll
      for (int j = 0; j < 8; j++) Bs[buf][(n0 + j) * LDS_STRIDE + k] = tmp[j];
    }
  };
#if USE_TDM
  // Tensor Data Mover: 2D tile 64(k) x 128(m) of 2-byte elems, row stride = K,
  // LDS padding 4 dwords (8 elems) after every 32 dwords (64 elems) -> stride 72.
  auto tdm_loadA = [&](int k0, int buf) {
    u32 lds_off = (u32)(uintptr_t)&As[buf][0];
    u64 ga = (u64)(uintptr_t)(Aptr + k0);
    v4u g0;
    g0[0] = 1u;                                    // count=1, user descriptor
    g0[1] = lds_off;                               // lds_addr
    g0[2] = (u32)ga;                               // global_addr[31:0]
    g0[3] = (u32)((ga >> 32) & 0x01FFFFFFull) | (2u << 30);  // addr[56:32], type=2
    v8i g1;
    g1[0] = (1 << 16) | (1 << 20) | (4 << 22) | (3 << 25);   // 2B, pad_en, 32dw, 4dw
    g1[1] = (int)((K & 0xFFFF) << 16);                       // tensor_dim0[15:0]
    g1[2] = (int)(((K >> 16) & 0xFFFF) | ((M & 0xFFFF) << 16));
    g1[3] = (int)(((M >> 16) & 0xFFFF) | (64u << 16));       // tile_dim0 = 64 (k)
    g1[4] = 128;                                             // tile_dim1 = 128 (m)
    g1[5] = (int)K;                                          // dim0 stride lo32
    g1[6] = 0; g1[7] = 0;
    TDM_LOAD(g0, g1);
  };
#else
  auto loadA = [&](int k0) {                       // A: 128(m) x 64(k)
#pragma unroll
    for (int i = 0; i < 4; i++) {
      int slot = tid + i * 256;                    // 0..1023 chunks
      int r = slot >> 3, c = slot & 7;
      ar[i] = *(const uint4*)(Aptr + (size_t)r * K + k0 + c * 8);
    }
  };
  auto storeA = [&](int buf) {
#pragma unroll
    for (int i = 0; i < 4; i++) {
      int slot = tid + i * 256;
      int r = slot >> 3, c = slot & 7;
      *(uint4*)&As[buf][r * LDS_STRIDE + c * 8] = ar[i];
    }
  };
#endif

  const int nk = K >> 6;
  // ---- prologue: stage tile 0 ----
#if USE_TDM
  if (tid < 32) tdm_loadA(0, 0);
#else
  loadA(0);
#endif
  loadB(0);
#if !USE_TDM
  storeA(0);
#endif
  storeB(0);
#if USE_TDM
  if (tid < 32) __builtin_amdgcn_s_wait_tensorcnt(0);
#endif
  __syncthreads();

  for (int t = 0; t < nk; ++t) {
    const int cur = t & 1, nxt = cur ^ 1;
    const bool more = (t + 1 < nk);
    if (more) {
#if USE_TDM
      if (tid < 32) tdm_loadA((t + 1) * 64, nxt);
#else
      loadA((t + 1) * 64);
#endif
      loadB((t + 1) * 64);
      __builtin_prefetch(Bptr + (size_t)((t + 1) * 64 + (tid >> 2)) * N, 0, 0);
    }
#pragma unroll
    for (int ks = 0; ks < 2; ks++) {
      bf16x16 af[2], bfr[4];
#pragma unroll
      for (int r = 0; r < 2; r++) af[r]  = frag_from_lds(As[cur], wr * 32 + r * 16, ks * 32);
#pragma unroll
      for (int c = 0; c < 4; c++) bfr[c] = frag_from_lds(Bs[cur], wc * 64 + c * 16, ks * 32);
#pragma unroll
      for (int r = 0; r < 2; r++)
#pragma unroll
        for (int c = 0; c < 4; c++)
          acc[r][c] = __builtin_amdgcn_wmma_f32_16x16x32_bf16(
              false, af[r], false, bfr[c], (short)0, acc[r][c], false, false);
    }
    if (more) {
#if !USE_TDM
      storeA(nxt);
#endif
      storeB(nxt);
#if USE_TDM
      if (tid < 32) __builtin_amdgcn_s_wait_tensorcnt(0);
#endif
      __syncthreads();
    }
  }

  // epilogue: C/D layout — lane 0-15: M=v, N=lane; lane 16-31: M=v+8, N=lane-16
  const int lane  = tid & 31;
  const int lhalf = lane >> 4;
  const int lcol  = lane & 15;
#pragma unroll
  for (int r = 0; r < 2; r++)
#pragma unroll
    for (int c = 0; c < 4; c++)
#pragma unroll
      for (int v = 0; v < 8; v++) {
        int row = tileM + wr * 32 + r * 16 + v + lhalf * 8;
        int col = tileN + wc * 64 + c * 16 + lcol;
        float val = acc[r][c][v] + bias[col];
        if (mode == 1) val = 0.5f * val * (1.0f + erff(val * 0.70710678118f));
        if (res) val += res[(size_t)row * N + col];
        C[(size_t)row * N + col] = val;
      }
}

// ---------- value_gates = sigmoid(gh @ Wg2 + bg2), one wave per row ----------
__global__ __launch_bounds__(128)
void rowdot_sigmoid(const float* __restrict__ gh, const float* __restrict__ w,
                    const float* __restrict__ b, float* __restrict__ vg) {
  int wid = threadIdx.x >> 5, lane = threadIdx.x & 31;
  int row = blockIdx.x * 4 + wid;
  const float* r = gh + (size_t)row * DQ;
  float s = 0.f;
  for (int d = lane; d < DQ; d += 32) s += r[d] * w[d];
  s = wave_sum(s);
  if (lane == 0) vg[row] = 1.0f / (1.0f + __expf(-(s + b[0])));
}

// ---------- bw = softmax(bh @ Wb2 + bb2), one wave per row ----------
__global__ __launch_bounds__(128)
void rowsoftmax2(const float* __restrict__ bh, const float* __restrict__ w,
                 const float* __restrict__ b2, float* __restrict__ bw) {
  int wid = threadIdx.x >> 5, lane = threadIdx.x & 31;
  int row = blockIdx.x * 4 + wid;
  const float* r = bh + (size_t)row * (DQ / 2);
  float s0 = 0.f, s1 = 0.f;
  for (int d = lane; d < DQ / 2; d += 32) {
    float v = r[d];
    s0 += v * w[d * 2 + 0];
    s1 += v * w[d * 2 + 1];
  }
  s0 = wave_sum(s0); s1 = wave_sum(s1);
  if (lane == 0) {
    float z0 = s0 + b2[0], z1 = s1 + b2[1];
    float m = fmaxf(z0, z1);
    float e0 = __expf(z0 - m), e1 = __expf(z1 - m);
    float inv = 1.0f / (e0 + e1);
    bw[row * 2 + 0] = e0 * inv;
    bw[row * 2 + 1] = e1 * inv;
  }
}

// ---------- gate cumsum (scalar per (b,l)), clamp >= 1 ----------
__global__ void gate_cumsum_kernel(const float* __restrict__ vg, float* __restrict__ gc) {
  int b = threadIdx.x;
  if (b >= BQ) return;
  float run = 0.f;
  for (int l = 0; l < LQ; ++l) {
    run += vg[b * LQ + l];
    gc[b * LQ + l] = fmaxf(run, 1.0f);
  }
}

// ---------- segmented scan pass 1: per-segment partial sums ----------
__global__ void scan_pass1(const float* __restrict__ pos_phases,
                           const float* __restrict__ pv,
                           const float* __restrict__ kpre,
                           const float* __restrict__ kvv,
                           const float* __restrict__ vg,
                           float* __restrict__ part) {
  int t  = blockIdx.x * 256 + threadIdx.x;
  int ch = t % (BQ * DQ);
  int s  = t / (BQ * DQ);
  int b = ch / DQ, d = ch % DQ;
  int l0 = s * SEGL;
  const float PIF = 3.14159265358979f;
  float kcp = 0.f, ksp = 0.f;
  if (l0 > 0) {
    float kph = tanhf(kpre[((size_t)(b * LQ + l0 - 1)) * DQ + d]) * PIF;
    kcp = __cosf(kph); ksp = __sinf(kph);
  }
  float smc = 0, sms = 0, skc = 0, sks = 0;
  for (int l = l0; l < l0 + SEGL; ++l) {
    size_t idx = ((size_t)(b * LQ + l)) * DQ + d;
    float pp = pos_phases[(size_t)l * DQ + d];
    float c = __cosf(pp), sn = __sinf(pp);
    float p = pv[idx];
    smc += c * p; sms += sn * p;
    float kph = tanhf(kpre[idx]) * PIF;
    float kc = __cosf(kph), ks = __sinf(kph);
    float kv = kvv[idx];
    float g  = vg[b * LQ + l];
    skc += kcp * kv * g; sks += ksp * kv * g;
    kcp = kc; ksp = ks;
  }
  float* o = part + ((size_t)ch * NSEG + s) * 4;
  o[0] = smc; o[1] = sms; o[2] = skc; o[3] = sks;
}

// ---------- exclusive prefix over segments per chain ----------
__global__ void scan_prefix(float* __restrict__ part) {
  int ch = blockIdx.x * 256 + threadIdx.x;
  if (ch >= BQ * DQ) return;
#pragma unroll
  for (int q = 0; q < 4; q++) {
    float run = 0.f;
    for (int s = 0; s < NSEG; s++) {
      float* p = part + ((size_t)ch * NSEG + s) * 4 + q;
      float t = *p; *p = run; run += t;
    }
  }
}

// ---------- segmented scan pass 2: emit blended (in-place over pv) ----------
__global__ void scan_pass2(const float* __restrict__ pos_phases,
                           const float* __restrict__ pv,
                           const float* __restrict__ kpre,
                           const float* __restrict__ kvv,
                           const float* __restrict__ vg,
                           const float* __restrict__ gc,
                           const float* __restrict__ bwp,
                           const float* __restrict__ part,
                           float* __restrict__ blended) {
  int t  = blockIdx.x * 256 + threadIdx.x;
  int ch = t % (BQ * DQ);
  int s  = t / (BQ * DQ);
  int b = ch / DQ, d = ch % DQ;
  int l0 = s * SEGL;
  const float PIF = 3.14159265358979f;
  const float invSqrtD = 0.03125f;  // 1/sqrt(1024)
  const float* off = part + ((size_t)ch * NSEG + s) * 4;
  float mc = off[0], ms = off[1], kvc = off[2], kvs = off[3];
  float kcp = 0.f, ksp = 0.f;
  if (l0 > 0) {
    float kph = tanhf(kpre[((size_t)(b * LQ + l0 - 1)) * DQ + d]) * PIF;
    kcp = __cosf(kph); ksp = __sinf(kph);
  }
  for (int l = l0; l < l0 + SEGL; ++l) {
    size_t idx = ((size_t)(b * LQ + l)) * DQ + d;
    float pp = pos_phases[(size_t)l * DQ + d];
    float c = __cosf(pp), sn = __sinf(pp);
    float p = pv[idx];
    mc += c * p; ms += sn * p;
    float pret = (c * mc + sn * ms) * invSqrtD;
    float kph = tanhf(kpre[idx]) * PIF;
    float kc = __cosf(kph), ks = __sinf(kph);
    float kv = kvv[idx];
    float g  = vg[b * LQ + l];
    kvc += kcp * kv * g; kvs += ksp * kv * g;
    float kret = (kc * kvc + ks * kvs) * rsqrtf(gc[b * LQ + l]) * invSqrtD;
    float w0 = bwp[(b * LQ + l) * 2 + 0], w1 = bwp[(b * LQ + l) * 2 + 1];
    blended[idx] = w0 * pret + w1 * kret;
    kcp = kc; ksp = ks;
  }
}

// ---------- LayerNorm row-wise, emit bf16 for final WMMA GEMM ----------
__global__ __launch_bounds__(128)
void layernorm_kernel(const float* __restrict__ bl, const float* __restrict__ g,
                      const float* __restrict__ be, u16* __restrict__ lnb) {
  int wid = threadIdx.x >> 5, lane = threadIdx.x & 31;
  int row = blockIdx.x * 4 + wid;
  const float* r = bl + (size_t)row * DQ;
  float lv[DQ / 32];
  float s = 0.f, s2 = 0.f;
#pragma unroll
  for (int i = 0; i < DQ / 32; i++) {
    float v = r[i * 32 + lane];
    lv[i] = v; s += v; s2 += v * v;
  }
  s = wave_sum(s); s2 = wave_sum(s2);
  float mu  = s * (1.0f / DQ);
  float var = s2 * (1.0f / DQ) - mu * mu;
  float inv = rsqrtf(var + 1e-5f);
#pragma unroll
  for (int i = 0; i < DQ / 32; i++) {
    int d = i * 32 + lane;
    float o = (lv[i] - mu) * inv * g[d] + be[d];
    lnb[(size_t)row * DQ + d] = f2bf(o);
  }
}

// ====================== host launcher ======================
extern "C" void kernel_launch(void* const* d_in, const int* in_sizes, int n_in,
                              void* d_out, int out_size, void* d_ws, size_t ws_size,
                              hipStream_t stream) {
  (void)in_sizes; (void)n_in; (void)out_size; (void)ws_size;
  const float* x   = (const float*)d_in[0];
  const float* pph = (const float*)d_in[1];
  const float* Wpv = (const float*)d_in[2];  const float* bpv = (const float*)d_in[3];
  const float* Wk  = (const float*)d_in[4];  const float* bk  = (const float*)d_in[5];
  const float* Wkv = (const float*)d_in[6];  const float* bkv = (const float*)d_in[7];
  const float* Wg1 = (const float*)d_in[8];  const float* bg1 = (const float*)d_in[9];
  const float* Wg2 = (const float*)d_in[10]; const float* bg2 = (const float*)d_in[11];
  const float* Wb1 = (const float*)d_in[12]; const float* bb1 = (const float*)d_in[13];
  const float* Wb2 = (const float*)d_in[14]; const float* bb2 = (const float*)d_in[15];
  const float* lng = (const float*)d_in[16]; const float* lnbias = (const float*)d_in[17];
  const float* Wo  = (const float*)d_in[18]; const float* bo  = (const float*)d_in[19];
  float* out = (float*)d_out;

  const int M = MQ, D = DQ, K2 = 2 * DQ, Dh = DQ / 2;

  char* ws = (char*)d_ws;
  size_t off = 0;
  auto take = [&](size_t bytes) -> char* {
    char* p = ws + off;
    off = (off + bytes + 255) & ~(size_t)255;
    return p;
  };
  u16* xb    = (u16*)take((size_t)M * D * 2);
  u16* wpv_b = (u16*)take((size_t)D * D * 2);
  u16* wk_b  = (u16*)take((size_t)D * D * 2);
  u16* wkv_b = (u16*)take((size_t)D * D * 2);
  u16* wo_b  = (u16*)take((size_t)D * D * 2);
  u16* wg1_b = (u16*)take((size_t)K2 * D * 2);
  u16* wb1_b = (u16*)take((size_t)D * Dh * 2);
  char* bufA = take((size_t)M * K2 * 2);   // gate_in bf16 -> pos_values f32 -> blended f32
  char* bufG = take((size_t)M * D * 4);    // gh f32 -> kpre f32
  float* kvv = (float*)take((size_t)M * D * 4);
  char* bufB = take((size_t)M * Dh * 4);   // bh f32 -> ln bf16
  float* vg  = (float*)take((size_t)M * 4);
  float* gc  = (float*)take((size_t)M * 4);
  float* bw  = (float*)take((size_t)M * 2 * 4);
  float* part = (float*)take((size_t)BQ * D * NSEG * 4 * 4);

  // 1) conversions to bf16 (4 elems/thread)
  cvt_f32_bf16<<<(M * D / 4 + 255) / 256, 256, 0, stream>>>(x, xb, M * D / 4);
  cvt_f32_bf16<<<(D * D / 4 + 255) / 256, 256, 0, stream>>>(Wpv, wpv_b, D * D / 4);
  cvt_f32_bf16<<<(D * D / 4 + 255) / 256, 256, 0, stream>>>(Wk, wk_b, D * D / 4);
  cvt_f32_bf16<<<(D * D / 4 + 255) / 256, 256, 0, stream>>>(Wkv, wkv_b, D * D / 4);
  cvt_f32_bf16<<<(D * D / 4 + 255) / 256, 256, 0, stream>>>(Wo, wo_b, D * D / 4);
  cvt_f32_bf16<<<(K2 * D / 4 + 255) / 256, 256, 0, stream>>>(Wg1, wg1_b, K2 * D / 4);
  cvt_f32_bf16<<<(D * Dh / 4 + 255) / 256, 256, 0, stream>>>(Wb1, wb1_b, D * Dh / 4);

  // 2) gate branch first (lets us reuse buffers)
  build_gate_in<<<(size_t)M * K2 / 8 / 256, 256, 0, stream>>>(xb, (u16*)bufA);
  gemm_bf16_wmma<<<dim3(D / 128, M / 128), 256, 0, stream>>>(
      (u16*)bufA, wg1_b, bg1, nullptr, (float*)bufG, M, D, K2, 1);          // gh
  rowdot_sigmoid<<<M / 4, 128, 0, stream>>>((float*)bufG, Wg2, bg2, vg);
  gate_cumsum_kernel<<<1, BQ, 0, stream>>>(vg, gc);

  // 3) linear projections (gate_in dead -> bufA = pos_values; gh dead -> bufG = kpre)
  gemm_bf16_wmma<<<dim3(D / 128, M / 128), 256, 0, stream>>>(
      xb, wpv_b, bpv, nullptr, (float*)bufA, M, D, D, 0);                   // pos_values
  gemm_bf16_wmma<<<dim3(D / 128, M / 128), 256, 0, stream>>>(
      xb, wk_b, bk, nullptr, (float*)bufG, M, D, D, 0);                     // kpre
  gemm_bf16_wmma<<<dim3(D / 128, M / 128), 256, 0, stream>>>(
      xb, wkv_b, bkv, nullptr, kvv, M, D, D, 0);                            // kv_values
  gemm_bf16_wmma<<<dim3(Dh / 128, M / 128), 256, 0, stream>>>(
      xb, wb1_b, bb1, nullptr, (float*)bufB, M, Dh, D, 1);                  // bh
  rowsoftmax2<<<M / 4, 128, 0, stream>>>((float*)bufB, Wb2, bb2, bw);

  // 4) segmented scans -> blended (in-place over pos_values)
  scan_pass1<<<(BQ * D * NSEG) / 256, 256, 0, stream>>>(
      pph, (float*)bufA, (float*)bufG, kvv, vg, part);
  scan_prefix<<<(BQ * D + 255) / 256, 256, 0, stream>>>(part);
  scan_pass2<<<(BQ * D * NSEG) / 256, 256, 0, stream>>>(
      pph, (float*)bufA, (float*)bufG, kvv, vg, gc, bw, part, (float*)bufA);

  // 5) LayerNorm -> bf16, final GEMM with residual into d_out
  layernorm_kernel<<<M / 4, 128, 0, stream>>>((float*)bufA, lng, lnbias, (u16*)bufB);
  gemm_bf16_wmma<<<dim3(D / 128, M / 128), 256, 0, stream>>>(
      (u16*)bufB, wo_b, bo, x, out, M, D, D, 0);
}
